// HieraAttLayer_24584392802390
// MI455X (gfx1250) — matrix operand order repaired and checked by reading, not compiled, not run on hardware
//
#include <hip/hip_runtime.h>
#include <hip/hip_bf16.h>
#include <math.h>

typedef __attribute__((ext_vector_type(16))) _Float16 v16h;
typedef __attribute__((ext_vector_type(8)))  _Float16 h8;
typedef __attribute__((ext_vector_type(8)))  float    v8f;

#define DEV __device__ __forceinline__

// ---------------- weight blob offsets (f16 elements) in workspace ----------------
// Stored COLUMN-MAJOR: blob[c * strideK + r], strideK = Kpad + 8 (bank-spread pad),
// rows >= K zero-filled. Contiguous per column -> ds_load_b128 B-fragments.
// Edge-MLP W0 blobs are ROW-ROTATED to match the rotated X_e column layout:
//   X_e cols: [X_n 0..127 | ft0_nn 128..191 | nrm 192..255 | ps 256..319 | pn 320..383 | d 384]
//   (original concat order was [d | X_n | ft0_nn | nrm | ps | pn])
constexpr int OFF_ET0_W0 = 0;           // 64 x 424  (385->416 pad ->424 stride)
constexpr int OFF_ET0_W1 = 27136;       // 64 x 72
constexpr int OFF_ET0_W2 = 31744;       // 16 x 72
constexpr int OFF_ET1_W0 = 32896;       // 64 x 424
constexpr int OFF_ET1_W1 = 60032;       // 64 x 72
constexpr int OFF_ET1_W2 = 64640;       // 48 x 72
constexpr int OFF_EVM_W0 = 68096;       // 128 x 424
constexpr int OFF_EVM_W1 = 122368;      // 128 x 136
constexpr int OFF_EVM_W2 = 139776;      // 128 x 136
constexpr int OFF_VQ_W0  = 157184;      // 64 x 136
constexpr int OFF_VQ_W1  = 165888;      // 64 x 72
constexpr int OFF_VQ_W2  = 170496;      // 256 x 72
constexpr int OFF_O0_W0  = 188928;      // 64 x 520
constexpr int OFF_O0_W1  = 222208;      // 64 x 72
constexpr int OFF_O0_W2  = 226816;      // 64 x 72
constexpr int OFF_T2_W   = 231424;      // 64 x 520
constexpr int OFF_WT_END = 264704;

// ---------------- WMMA fragment helpers (CDNA5 16x16x32 f16, wave32) ----------------
DEV v16h load_a_frag(const _Float16* A, int ldA, int row0, int kb) {
  int lane = threadIdx.x & 31;
  int half = lane >> 4, r = lane & 15;
  const _Float16* p0 = A + (row0 + r) * ldA + kb + half * 16;
  v16h a;
#pragma unroll
  for (int j = 0; j < 8; ++j) a[j] = p0[j];
#pragma unroll
  for (int j = 0; j < 8; ++j) a[8 + j] = p0[8 + j];
  return a;
}

DEV v16h load_b_frag(const _Float16* W, int Wld, int col0, int kb) {
  int lane = threadIdx.x & 31;
  int half = lane >> 4, c = lane & 15;
  const _Float16* p = W + (col0 + c) * Wld + kb + half * 16;
  v16h b;
#pragma unroll
  for (int j = 0; j < 16; ++j) b[j] = p[j];
  return b;
}

// Software-pipelined: next fragments are in flight while current WMMA executes,
// so the dscnt wait before each WMMA tolerates outstanding loads.
DEV void gemm_acc(v8f& acc, const _Float16* A, int ldA, int row0,
                  const _Float16* W, int Wld, int col0, int K) {
  v16h a = load_a_frag(A, ldA, row0, 0);
  v16h b = load_b_frag(W, Wld, col0, 0);
  for (int kb = 32; kb < K; kb += 32) {
    v16h an = load_a_frag(A, ldA, row0, kb);
    v16h bn = load_b_frag(W, Wld, col0, kb);
    acc = __builtin_amdgcn_wmma_f32_16x16x32_f16(false, a, false, b,
                                                 (short)0, acc, false, false);
    a = an; b = bn;
  }
  acc = __builtin_amdgcn_wmma_f32_16x16x32_f16(false, a, false, b,
                                               (short)0, acc, false, false);
}

DEV float elu1(float x) { return x > 0.f ? x : __expf(x) - 1.f; }

DEV void store_tile_h16(_Float16* H, int ldH, int row0, int col0, const v8f& acc,
                        const float* bias, bool act) {
  int lane = threadIdx.x & 31, half = lane >> 4, c = lane & 15;
#pragma unroll
  for (int i = 0; i < 8; ++i) {
    float v = acc[i] + bias[col0 + c];
    if (act) v = elu1(v);
    H[(row0 + i + 8 * half) * ldH + col0 + c] = (_Float16)v;
  }
}

DEV void store_tile_f32(float* H, int ldH, int row0, int col0, const v8f& acc,
                        const float* bias) {
  int lane = threadIdx.x & 31, half = lane >> 4, c = lane & 15;
#pragma unroll
  for (int i = 0; i < 8; ++i)
    H[(row0 + i + 8 * half) * ldH + col0 + c] = acc[i] + bias[col0 + c];
}

DEV void mlp_layer(const _Float16* A, int ldA, int K, const _Float16* W, int Wld,
                   int Ncol, const float* bias, bool act, _Float16* Hout, int ldH) {
  int w = threadIdx.x >> 5;
  int ntiles = 2 * (Ncol >> 4);
  for (int t = w; t < ntiles; t += 8) {
    int mt = t & 1, nt = t >> 1;
    v8f acc = {};
    gemm_acc(acc, A, ldA, mt * 16, W, Wld, nt * 16, K);
    store_tile_h16(Hout, ldH, mt * 16, nt * 16, acc, bias, act);
  }
}

DEV void mlp_layer_f32(const _Float16* A, int ldA, int K, const _Float16* W, int Wld,
                       int Ncol, const float* bias, float* Hout, int ldH) {
  int w = threadIdx.x >> 5;
  int ntiles = 2 * (Ncol >> 4);
  for (int t = w; t < ntiles; t += 8) {
    int mt = t & 1, nt = t >> 1;
    v8f acc = {};
    gemm_acc(acc, A, ldA, mt * 16, W, Wld, nt * 16, K);
    store_tile_f32(Hout, ldH, mt * 16, nt * 16, acc, bias);
  }
}

// 128-bit cooperative copy; n must be a multiple of 8 halves, 16B-aligned ptrs.
DEV void copy_h16(_Float16* dst, const _Float16* src, int n) {
  uint4* d = (uint4*)dst;
  const uint4* s = (const uint4*)src;
  int n16 = n >> 3;
  for (int i = threadIdx.x; i < n16; i += 256) d[i] = s[i];
}

// ---------------- prep kernels ----------------
// dst column-major: dst[c*strideK + r] = src[srcRow(r)*Ncol + c], zero for r >= K.
// rot=1 applies the X_e column rotation: dst row r <- src row (r+1), dst row K-1 <- src row 0.
__global__ __launch_bounds__(256) void convert_pad_kernel(_Float16* dst, const float* src,
                                                          int K, int Ncol, int strideK,
                                                          int rot) {
  int tot = strideK * Ncol;
  for (int i = blockIdx.x * blockDim.x + threadIdx.x; i < tot; i += gridDim.x * blockDim.x) {
    int c = i / strideK, r = i - c * strideK;
    int rr = r;
    if (rot) rr = (r < K - 1) ? (r + 1) : ((r == K - 1) ? 0 : K);
    dst[i] = (rr < K) ? (_Float16)src[rr * Ncol + c] : (_Float16)0.f;
  }
}

__global__ __launch_bounds__(256) void prep_xn_kernel(_Float16* xn, const float* ft0,
                                                      const float* ft1, int N) {
  int tot = N * 128;
  for (int i = blockIdx.x * blockDim.x + threadIdx.x; i < tot; i += gridDim.x * blockDim.x) {
    int n = i >> 7, c = i & 127;
    float v;
    if (c < 64) {
      v = ft0[n * 64 + c];
    } else {
      int s = c - 64;
      float a = ft1[n * 192 + s], b = ft1[n * 192 + 64 + s], d = ft1[n * 192 + 128 + s];
      v = sqrtf(a * a + b * b + d * d);
    }
    xn[i] = (_Float16)v;
  }
}

// ---------------- Q = vt0emb MLP (128->64->64->256), 32 nodes per block ----------------
__global__ __launch_bounds__(256) void q_kernel(const _Float16* xn, const _Float16* wts,
                                                const float* b0, const float* b1,
                                                const float* b2, float* Qout, int N) {
  extern __shared__ unsigned char smem[];
  _Float16* sX  = (_Float16*)smem;  // 32*136
  _Float16* sW  = sX + 4352;        // up to 256*72 = 18432
  _Float16* sH1 = sW + 18432;       // 32*72
  _Float16* sH2 = sH1 + 2304;       // 32*72

  int base = blockIdx.x * 32;
  for (int i = threadIdx.x; i < 32 * 16; i += 256) {
    int r = i >> 4, c8 = (i & 15) * 8;
    int node = base + r;
    h8 v = {};
    if (node < N) v = *(const h8*)(xn + (size_t)node * 128 + c8);
    *(h8*)(sX + r * 136 + c8) = v;
  }
  copy_h16(sW, wts + OFF_VQ_W0, 64 * 136);
  __syncthreads();
  mlp_layer(sX, 136, 128, sW, 136, 64, b0, true, sH1, 72);
  __syncthreads();
  copy_h16(sW, wts + OFF_VQ_W1, 64 * 72);
  __syncthreads();
  mlp_layer(sH1, 72, 64, sW, 72, 64, b1, true, sH2, 72);
  __syncthreads();
  copy_h16(sW, wts + OFF_VQ_W2, 256 * 72);
  __syncthreads();
  int w = threadIdx.x >> 5;
  for (int t = w; t < 32; t += 8) {  // 2 x 16 tiles
    int mt = t & 1, nt = t >> 1;
    v8f acc = {};
    gemm_acc(acc, sH2, 72, mt * 16, sW, 72, nt * 16, 64);
    int lane = threadIdx.x & 31, half = lane >> 4, c = lane & 15;
#pragma unroll
    for (int i = 0; i < 8; ++i) {
      int node = base + mt * 16 + i + 8 * half;
      if (node < N) Qout[(size_t)node * 256 + nt * 16 + c] = acc[i] + b2[nt * 16 + c];
    }
  }
}

// ---------------- fused per-node kernel: gather + edge MLPs + attention ----------------
__global__ __launch_bounds__(256) void node_kernel(
    const float* ft0, const float* ft1, const int* ids, const float* dtk, const float* rtk,
    const _Float16* wts, const _Float16* xn, const float* Q,
    const float* b_e00, const float* b_e01, const float* b_e02,
    const float* b_e10, const float* b_e11, const float* b_e12,
    const float* b_v0,  const float* b_v1,  const float* b_v2,
    _Float16* ut0, _Float16* ut1, int N) {
  extern __shared__ unsigned char smem[];
  _Float16* sXe = (_Float16*)smem;   // 32*424 = 13568
  _Float16* sW  = sXe + 13568;       // 64*424 = 27136 (max staged weight slice)
  _Float16* sH1 = sW + 27136;        // 32*136
  _Float16* sH2 = sH1 + 4352;        // 32*136
  _Float16* sXn = sH2 + 4352;        // 128
  float* sV   = (float*)(sXn + 128); // 32*128
  float* sKt0 = sV + 4096;           // 32*16
  float* sKt1 = sKt0 + 512;          // 32*48
  float* sF1  = sKt1 + 1536;         // 32*192 (gathered ft1_nn)
  float* sFt1 = sF1 + 6144;          // 3*64 (self ft1)
  float* sR   = sFt1 + 192;          // 32*3
  float* sD   = sR + 96;             // 32
  float* sQ   = sD + 32;             // 256
  float* sM0  = sQ + 256;            // 8*32
  float* sM1  = sM0 + 256;           // 8*96
  float* sMid = sM1 + 768;           // 8
  int*   sId  = (int*)(sMid + 8);    // 32

  const int n = blockIdx.x;
  const int tid = threadIdx.x;

  // ---- phase 0: node-local loads ----
  for (int i = tid; i < 192; i += 256) sFt1[i] = ft1[(size_t)n * 192 + i];
  for (int i = tid; i < 96; i += 256) sR[i] = rtk[(size_t)n * 120 + i];
  for (int i = tid; i < 32; i += 256) sD[i] = dtk[(size_t)n * 40 + i];
  for (int i = tid; i < 256; i += 256) sQ[i] = Q[(size_t)n * 256 + i];
  for (int i = tid; i < 128; i += 256) sXn[i] = xn[(size_t)n * 128 + i];
  if (tid < 32) {
    int g = ids[(size_t)n * 40 + tid];
    sId[tid] = g;
    __builtin_prefetch(ft0 + (size_t)g * 64, 0, 3);
    __builtin_prefetch(ft1 + (size_t)g * 192, 0, 3);
    __builtin_prefetch(ft1 + (size_t)g * 192 + 96, 0, 3);
  }
  __syncthreads();

  // ---- phase 1: gather neighbors, build X_e (rotated layout, 32 x 424 f16) ----
  {
    int k = tid >> 3, sub = tid & 7, s0 = sub * 8;
    int g = sId[k];
    float r0 = sR[k * 3], r1 = sR[k * 3 + 1], r2 = sR[k * 3 + 2];
    const float* f1g = ft1 + (size_t)g * 192;
    const float* f0g = ft0 + (size_t)g * 64;
    float va[3][8];
#pragma unroll
    for (int d = 0; d < 3; ++d) {
      float4 lo = *(const float4*)(f1g + d * 64 + s0);
      float4 hi = *(const float4*)(f1g + d * 64 + s0 + 4);
      va[d][0] = lo.x; va[d][1] = lo.y; va[d][2] = lo.z; va[d][3] = lo.w;
      va[d][4] = hi.x; va[d][5] = hi.y; va[d][6] = hi.z; va[d][7] = hi.w;
      *(float4*)(sF1 + k * 192 + d * 64 + s0) = lo;
      *(float4*)(sF1 + k * 192 + d * 64 + s0 + 4) = hi;
    }
    float4 flo = *(const float4*)(f0g + s0);
    float4 fhi = *(const float4*)(f0g + s0 + 4);
    float f0v[8] = {flo.x, flo.y, flo.z, flo.w, fhi.x, fhi.y, fhi.z, fhi.w};
    h8 hf0, hnrm, hps, hpn;
#pragma unroll
    for (int j = 0; j < 8; ++j) {
      float A0 = va[0][j], A1 = va[1][j], A2 = va[2][j];
      hnrm[j] = (_Float16)sqrtf(A0 * A0 + A1 * A1 + A2 * A2);
      hpn[j]  = (_Float16)(A0 * r0 + A1 * r1 + A2 * r2);
      int s = s0 + j;
      hps[j]  = (_Float16)(sFt1[s] * r0 + sFt1[64 + s] * r1 + sFt1[128 + s] * r2);
      hf0[j]  = (_Float16)f0v[j];
    }
    _Float16* row = sXe + k * 424;
    *(h8*)(row + 128 + s0) = hf0;
    *(h8*)(row + 192 + s0) = hnrm;
    *(h8*)(row + 256 + s0) = hps;
    *(h8*)(row + 320 + s0) = hpn;
  }
  for (int i = tid; i < 32 * 16; i += 256) {  // X_n cols 0..127
    int k = i >> 4, c8 = (i & 15) * 8;
    *(h8*)(sXe + k * 424 + c8) = *(const h8*)(sXn + c8);
  }
  for (int i = tid; i < 32 * 40; i += 256) {  // col 384 = d, 385..423 zero
    int k = i / 40, c = i - k * 40;
    sXe[k * 424 + 384 + c] = (c == 0) ? (_Float16)sD[k] : (_Float16)0.f;
  }
  __syncthreads();

  // ---- phase 2: et0emb 416->64->64->16 ----
  copy_h16(sW, wts + OFF_ET0_W0, 64 * 424);
  __syncthreads();
  mlp_layer(sXe, 424, 416, sW, 424, 64, b_e00, true, sH1, 72);
  __syncthreads();
  copy_h16(sW, wts + OFF_ET0_W1, 64 * 72);
  __syncthreads();
  mlp_layer(sH1, 72, 64, sW, 72, 64, b_e01, true, sH2, 72);
  __syncthreads();
  copy_h16(sW, wts + OFF_ET0_W2, 16 * 72);
  __syncthreads();
  mlp_layer_f32(sH2, 72, 64, sW, 72, 16, b_e02, sKt0, 16);
  __syncthreads();

  // ---- phase 3: et1emb 416->64->64->48 ----
  copy_h16(sW, wts + OFF_ET1_W0, 64 * 424);
  __syncthreads();
  mlp_layer(sXe, 424, 416, sW, 424, 64, b_e10, true, sH1, 72);
  __syncthreads();
  copy_h16(sW, wts + OFF_ET1_W1, 64 * 72);
  __syncthreads();
  mlp_layer(sH1, 72, 64, sW, 72, 64, b_e11, true, sH2, 72);
  __syncthreads();
  copy_h16(sW, wts + OFF_ET1_W2, 48 * 72);
  __syncthreads();
  mlp_layer_f32(sH2, 72, 64, sW, 72, 48, b_e12, sKt1, 48);
  __syncthreads();

  // ---- phase 4: evm 416->128->128->128, N in halves of 64 (column-major slices) ----
  for (int nh = 0; nh < 2; ++nh) {
    copy_h16(sW, wts + OFF_EVM_W0 + nh * 64 * 424, 64 * 424);
    __syncthreads();
    mlp_layer(sXe, 424, 416, sW, 424, 64, b_v0 + nh * 64, true, sH1 + nh * 64, 136);
    __syncthreads();
  }
  for (int nh = 0; nh < 2; ++nh) {
    copy_h16(sW, wts + OFF_EVM_W1 + nh * 64 * 136, 64 * 136);
    __syncthreads();
    mlp_layer(sH1, 136, 128, sW, 136, 64, b_v1 + nh * 64, true, sH2 + nh * 64, 136);
    __syncthreads();
  }
  for (int nh = 0; nh < 2; ++nh) {
    copy_h16(sW, wts + OFF_EVM_W2 + nh * 64 * 136, 64 * 136);
    __syncthreads();
    mlp_layer_f32(sH2, 136, 128, sW, 136, 64, b_v2 + nh * 64, sV + nh * 64, 128);
    __syncthreads();
  }

  // ---- phase 5: attention ----
  {  // scores t0: 8 heads x 32 keys
    int h = tid >> 5, e = tid & 31;
    float s = 0.f;
#pragma unroll
    for (int j = 0; j < 16; ++j) s += sQ[h * 16 + j] * sKt0[e * 16 + j];
    sM0[h * 32 + e] = s * 0.25f;
  }
  for (int i = tid; i < 768; i += 256) {  // scores t1: 8 x 96; Kt1[e=c*32+k] = et1[k, c*16:+16]
    int h = i / 96, e = i - h * 96;
    int c = e >> 5, k = e & 31;
    const float* kt = sKt1 + k * 48 + c * 16;
    float s = 0.f;
#pragma unroll
    for (int j = 0; j < 16; ++j) s += sQ[128 + h * 16 + j] * kt[j];
    sM1[h * 96 + e] = s * 0.25f;
  }
  __syncthreads();
  if (tid < 8) {  // softmax per head
    int h = tid;
    float mx = -1e30f;
    for (int e = 0; e < 32; ++e) mx = fmaxf(mx, sM0[h * 32 + e]);
    float sum = 0.f;
    for (int e = 0; e < 32; ++e) { float ex = __expf(sM0[h * 32 + e] - mx); sM0[h * 32 + e] = ex; sum += ex; }
    float inv = 1.f / sum;
    for (int e = 0; e < 32; ++e) sM0[h * 32 + e] *= inv;

    mx = -1e30f;
    for (int e = 0; e < 96; ++e) mx = fmaxf(mx, sM1[h * 96 + e]);
    sum = 0.f;
    for (int e = 0; e < 96; ++e) { float ex = __expf(sM1[h * 96 + e] - mx); sM1[h * 96 + e] = ex; sum += ex; }
    inv = 1.f / sum;
    float mid = 0.f;
    for (int e = 0; e < 96; ++e) { sM1[h * 96 + e] *= inv; if (e >= 32 && e < 64) mid += sM1[h * 96 + e]; }
    sMid[h] = mid;
  }
  __syncthreads();

  for (int o = tid; o < 512; o += 256) {  // Ut0 = Mt0 @ V0
    int h = o >> 6, s = o & 63;
    float u = 0.f;
#pragma unroll 4
    for (int k = 0; k < 32; ++k) u += sM0[h * 32 + k] * sV[k * 128 + s];
    ut0[(size_t)n * 512 + o] = (_Float16)u;
  }
  for (int o = tid; o < 1536; o += 256) {  // Ut1[d][h][s]
    int d = o >> 9, rem = o & 511, h = rem >> 6, s = rem & 63;
    float u = sMid[h] * sFt1[d * 64 + s];
#pragma unroll 4
    for (int k = 0; k < 32; ++k) {
      u += sM1[h * 96 + k] * sV[k * 128 + 64 + s] * sR[k * 3 + d];
      u += sM1[h * 96 + 64 + k] * sF1[k * 192 + d * 64 + s];
    }
    ut1[((size_t)n * 3 + d) * 512 + rem] = (_Float16)u;
  }
}

// ---------------- out0: ft0_u = ft0 + t0t1emb(Ut0), row 0 zeroed ----------------
__global__ __launch_bounds__(256) void out0_kernel(const _Float16* ut0, const _Float16* wts,
                                                   const float* b0, const float* b1,
                                                   const float* b2, const float* ft0,
                                                   float* out, int N) {
  extern __shared__ unsigned char smem[];
  _Float16* sX  = (_Float16*)smem;  // 32*520
  _Float16* sW  = sX + 16640;       // 64*520 = 33280 (full layer-1 weight)
  _Float16* sH1 = sW + 33280;       // 32*72
  _Float16* sH2 = sH1 + 2304;       // 32*72

  int base = blockIdx.x * 32;
  for (int i = threadIdx.x; i < 32 * 64; i += 256) {
    int r = i >> 6, c8 = (i & 63) * 8;
    int node = base + r;
    h8 v = {};
    if (node < N) v = *(const h8*)(ut0 + (size_t)node * 512 + c8);
    *(h8*)(sX + r * 520 + c8) = v;
  }
  copy_h16(sW, wts + OFF_O0_W0, 64 * 520);
  __syncthreads();
  int w = threadIdx.x >> 5, mt = w & 1, nt = w >> 1;
  v8f acc = {};
  gemm_acc(acc, sX, 520, mt * 16, sW, 520, nt * 16, 512);
  store_tile_h16(sH1, 72, mt * 16, nt * 16, acc, b0, true);
  __syncthreads();
  copy_h16(sW, wts + OFF_O0_W1, 64 * 72);
  __syncthreads();
  v8f acc2 = {};
  gemm_acc(acc2, sH1, 72, mt * 16, sW, 72, nt * 16, 64);
  store_tile_h16(sH2, 72, mt * 16, nt * 16, acc2, b1, true);
  __syncthreads();
  copy_h16(sW, wts + OFF_O0_W2, 64 * 72);
  __syncthreads();
  v8f acc3 = {};
  gemm_acc(acc3, sH2, 72, mt * 16, sW, 72, nt * 16, 64);
  int lane = threadIdx.x & 31, half = lane >> 4, c = lane & 15;
#pragma unroll
  for (int i = 0; i < 8; ++i) {
    int node = base + mt * 16 + i + 8 * half;
    if (node < N) {
      float v = acc3[i] + b2[nt * 16 + c] + ft0[(size_t)node * 64 + nt * 16 + c];
      if (node == 0) v = 0.f;
      out[(size_t)node * 64 + nt * 16 + c] = v;
    }
  }
}

// ---------------- out1: ft1_u = ft1 + Ut1 @ t2t2_W, node 0 zeroed ----------------
__global__ __launch_bounds__(256) void out1_kernel(const _Float16* ut1, const _Float16* wts,
                                                   const float* ft1, float* out, int R) {
  extern __shared__ unsigned char smem[];
  _Float16* sX = (_Float16*)smem;  // 32*520
  _Float16* sW = sX + 16640;       // 64*520 full weight

  int base = blockIdx.x * 32;
  for (int i = threadIdx.x; i < 32 * 64; i += 256) {
    int r = i >> 6, c8 = (i & 63) * 8;
    int row = base + r;
    h8 v = {};
    if (row < R) v = *(const h8*)(ut1 + (size_t)row * 512 + c8);
    *(h8*)(sX + r * 520 + c8) = v;
  }
  copy_h16(sW, wts + OFF_T2_W, 64 * 520);
  __syncthreads();
  int w = threadIdx.x >> 5, mt = w & 1, nt = w >> 1;
  v8f acc = {};
  gemm_acc(acc, sX, 520, mt * 16, sW, 520, nt * 16, 512);
  int lane = threadIdx.x & 31, half = lane >> 4, c = lane & 15;
#pragma unroll
  for (int i = 0; i < 8; ++i) {
    int row = base + mt * 16 + i + 8 * half;
    if (row < R) {
      float v = acc[i] + ft1[(size_t)row * 64 + nt * 16 + c];
      if (row < 3) v = 0.f;  // node 0 -> rows 0..2
      out[(size_t)row * 64 + nt * 16 + c] = v;
    }
  }
}

// ---------------- host launcher ----------------
extern "C" void kernel_launch(void* const* d_in, const int* in_sizes, int n_in,
                              void* d_out, int out_size, void* d_ws, size_t ws_size,
                              hipStream_t stream) {
  const float* ft0 = (const float*)d_in[0];
  const float* ft1 = (const float*)d_in[1];
  const int*   ids = (const int*)d_in[2];
  const float* dtk = (const float*)d_in[3];
  const float* rtk = (const float*)d_in[4];
  // params, insertion order: vt0emb, et0emb, et1emb, evm, t0t1emb, t2t2_W
  const float* vq_W0 = (const float*)d_in[5];
  const float* vq_b0 = (const float*)d_in[6];
  const float* vq_W1 = (const float*)d_in[7];
  const float* vq_b1 = (const float*)d_in[8];
  const float* vq_W2 = (const float*)d_in[9];
  const float* vq_b2 = (const float*)d_in[10];
  const float* e0_W0 = (const float*)d_in[11];
  const float* e0_b0 = (const float*)d_in[12];
  const float* e0_W1 = (const float*)d_in[13];
  const float* e0_b1 = (const float*)d_in[14];
  const float* e0_W2 = (const float*)d_in[15];
  const float* e0_b2 = (const float*)d_in[16];
  const float* e1_W0 = (const float*)d_in[17];
  const float* e1_b0 = (const float*)d_in[18];
  const float* e1_W1 = (const float*)d_in[19];
  const float* e1_b1 = (const float*)d_in[20];
  const float* e1_W2 = (const float*)d_in[21];
  const float* e1_b2 = (const float*)d_in[22];
  const float* ev_W0 = (const float*)d_in[23];
  const float* ev_b0 = (const float*)d_in[24];
  const float* ev_W1 = (const float*)d_in[25];
  const float* ev_b1 = (const float*)d_in[26];
  const float* ev_W2 = (const float*)d_in[27];
  const float* ev_b2 = (const float*)d_in[28];
  const float* o0_W0 = (const float*)d_in[29];
  const float* o0_b0 = (const float*)d_in[30];
  const float* o0_W1 = (const float*)d_in[31];
  const float* o0_b1 = (const float*)d_in[32];
  const float* o0_W2 = (const float*)d_in[33];
  const float* o0_b2 = (const float*)d_in[34];
  const float* t2_W  = (const float*)d_in[35];

  const int N = in_sizes[0] / 64;
  float* out = (float*)d_out;

  // workspace carve (f16 region then f32 Q)
  _Float16* wsH = (_Float16*)d_ws;
  size_t offXN  = OFF_WT_END;
  size_t offUT0 = offXN + (size_t)N * 128;
  size_t offUT1 = offUT0 + (size_t)N * 512;
  size_t offEnd = offUT1 + (size_t)N * 3 * 512;
  _Float16* xnB  = wsH + offXN;
  _Float16* ut0B = wsH + offUT0;
  _Float16* ut1B = wsH + offUT1;
  float* Qbuf = (float*)(wsH + ((offEnd + 1) & ~(size_t)1));

  // --- weight conversion to padded column-major f16 blobs ---
  struct WJob { int off; const float* src; int K, Ncol, strideK, rot; };
  const WJob jobs[16] = {
      {OFF_ET0_W0, e0_W0, 385, 64, 424, 1},  {OFF_ET0_W1, e0_W1, 64, 64, 72, 0},
      {OFF_ET0_W2, e0_W2, 64, 16, 72, 0},    {OFF_ET1_W0, e1_W0, 385, 64, 424, 1},
      {OFF_ET1_W1, e1_W1, 64, 64, 72, 0},    {OFF_ET1_W2, e1_W2, 64, 48, 72, 0},
      {OFF_EVM_W0, ev_W0, 385, 128, 424, 1}, {OFF_EVM_W1, ev_W1, 128, 128, 136, 0},
      {OFF_EVM_W2, ev_W2, 128, 128, 136, 0}, {OFF_VQ_W0, vq_W0, 128, 64, 136, 0},
      {OFF_VQ_W1, vq_W1, 64, 64, 72, 0},     {OFF_VQ_W2, vq_W2, 64, 256, 72, 0},
      {OFF_O0_W0, o0_W0, 512, 64, 520, 0},   {OFF_O0_W1, o0_W1, 64, 64, 72, 0},
      {OFF_O0_W2, o0_W2, 64, 64, 72, 0},     {OFF_T2_W, t2_W, 512, 64, 520, 0}};
  for (int j = 0; j < 16; ++j) {
    int tot = jobs[j].strideK * jobs[j].Ncol;
    int blocks = (tot + 255) / 256;
    if (blocks > 1024) blocks = 1024;
    convert_pad_kernel<<<blocks, 256, 0, stream>>>(wsH + jobs[j].off, jobs[j].src,
                                                   jobs[j].K, jobs[j].Ncol,
                                                   jobs[j].strideK, jobs[j].rot);
  }

  prep_xn_kernel<<<(N * 128 + 255) / 256, 256, 0, stream>>>(xnB, ft0, ft1, N);

  const int nTiles = (N + 31) / 32;
  q_kernel<<<nTiles, 256, 54784, stream>>>(xnB, wsH, vq_b0, vq_b1, vq_b2, Qbuf, N);

  node_kernel<<<N, 256, 154784, stream>>>(ft0, ft1, ids, dtk, rtk, wsH, xnB, Qbuf,
                                          e0_b0, e0_b1, e0_b2, e1_b0, e1_b1, e1_b2,
                                          ev_b0, ev_b1, ev_b2, ut0B, ut1B, N);

  out0_kernel<<<nTiles, 256, 109056, stream>>>(ut0B, wsH, o0_b0, o0_b1, o0_b2, ft0, out, N);

  const int R = 3 * N;
  out1_kernel<<<(R + 31) / 32, 256, 99840, stream>>>(ut1B, wsH, ft1, out + (size_t)N * 64, R);
}
